// FeatureQuantizer_60301340836058
// MI455X (gfx1250) — matrix operand order repaired and checked by reading, not compiled, not run on hardware
//
#include <hip/hip_runtime.h>
#include <hip/hip_bf16.h>

typedef __attribute__((ext_vector_type(16))) __bf16 v16bf;
typedef __attribute__((ext_vector_type(8)))  __bf16 v8bf;
typedef __attribute__((ext_vector_type(8)))  float  v8f;

// Problem constants
#define N_ROWS   65536      // B*H*W
#define C_DIM    256        // emb_dim
#define K_CODES  1024       // num_emb
#define HW       1024       // H*W
#define IMG_STR  262144     // C*H*W
#define A_STRIDE 264        // halfs per LDS A row (256 + 8 pad, keeps 16B align, 4-bank skew)
#define X_STRIDE 260        // floats per LDS x row (256 + 4 pad, 4-bank skew)
#define NWAVE    8

__device__ __forceinline__ unsigned short f32_to_bf16_bits(float v) {
    unsigned int u = __float_as_uint(v);
    unsigned int r = (u + 0x7FFFu + ((u >> 16) & 1u)) >> 16;   // round-to-nearest-even
    return (unsigned short)r;
}

// ---------------- Stage 0a: transposed bf16 codebook embT[code][c] ----------------
__global__ __launch_bounds__(256) void vq_prep_embT(const float* __restrict__ embed,
                                                    unsigned short* __restrict__ embT) {
    int g  = blockIdx.x * 256 + threadIdx.x;   // 32768 threads
    int k  = g & 1023;                         // code (lanes consecutive -> coalesced reads)
    int c0 = (g >> 10) * 8;
#pragma unroll
    for (int j = 0; j < 8; ++j) {
        float v = embed[(c0 + j) * K_CODES + k];
        embT[k * C_DIM + c0 + j] = f32_to_bf16_bits(v);
    }
}

// ---------------- Stage 0b: ||e_k||^2 ----------------
__global__ __launch_bounds__(256) void vq_prep_enorm(const float* __restrict__ embed,
                                                     float* __restrict__ enorm) {
    int k = blockIdx.x * 256 + threadIdx.x;    // 1024 threads
    float s = 0.0f;
    for (int c = 0; c < C_DIM; ++c) {
        float v = embed[c * K_CODES + k];
        s += v * v;
    }
    enorm[k] = s;
}

// ---------------- Stage 1: fused GEMM(WMMA bf16) + argmin + gather + loss ----------------
__global__ __launch_bounds__(256) void vq_stage1(const float* __restrict__ x,
                                                 const float* __restrict__ embed,
                                                 const unsigned short* __restrict__ embT,
                                                 const float* __restrict__ enorm,
                                                 float* __restrict__ out_q,
                                                 int* __restrict__ out_idx,
                                                 float* __restrict__ loss_partial) {
    __shared__ unsigned short a_lds[16 * A_STRIDE];   // bf16 x tile [row][c]
    __shared__ float          x_lds[16 * X_STRIDE];   // fp32 x tile [row][c]
    __shared__ float          red_val[16][NWAVE];
    __shared__ int            red_idx[16][NWAVE];
    __shared__ int            final_idx[16];
    __shared__ float          lred[NWAVE];

    const int t     = threadIdx.x;
    const int lane  = t & 31;
    const int wave  = t >> 5;
    const int nbase = blockIdx.x * 16;                // 16 rows per WG, same (b,h)
    const int xbase = (nbase >> 10) * IMG_STR + (nbase & 1023);  // + c*HW + row

    // ---- load x tile: coalesced (lanes 0..15 = consecutive rows / addresses) ----
    {
        const int row = t & 15;
        const int c0  = t >> 4;
#pragma unroll
        for (int i = 0; i < 16; ++i) {
            int c   = c0 + 16 * i;
            float v = x[xbase + c * HW + row];
            x_lds[row * X_STRIDE + c] = v;
            a_lds[row * A_STRIDE + c] = f32_to_bf16_bits(v);
        }
    }
    __syncthreads();

    const int half  = lane >> 4;      // lane group
    const int l16   = lane & 15;
    const int m     = l16;            // A-matrix row held by this lane
    const int akoff = half * 16;      // A layout: half 1 holds K+16..K+31
    const int bkoff = half * 16;      // B layout: half 1 holds K+16..K+31

    float best[8];
    int   bidx[8];
#pragma unroll
    for (int r = 0; r < 8; ++r) { best[r] = 3.4e38f; bidx[r] = 0; }

    // preload the 8 enorm values for this wave's code tiles (one clause, no in-loop waits)
    float en[8];
#pragma unroll
    for (int tile = 0; tile < 8; ++tile)
        en[tile] = enorm[wave * 128 + tile * 16 + l16];

    // ---- 8 code tiles (16 codes each) per wave; 8 WMMAs of 16x16x32 over C=256 ----
#pragma unroll
    for (int tile = 0; tile < 8; ++tile) {
        const int n0 = wave * 128 + tile * 16;
        const unsigned short* bptr = embT + (n0 + l16) * C_DIM + bkoff;

        // load ALL 8 B-fragments (8 x 32B) for this tile up front -> one clause, one wait
        v16bf bf[8];
#pragma unroll
        for (int i = 0; i < 8; ++i) {
            v8bf lo = *(const v8bf*)(bptr + i * 32);       // K = i*32+bkoff+0..7
            v8bf hi = *(const v8bf*)(bptr + i * 32 + 8);   // K = i*32+bkoff+8..15
            v16bf f;
#pragma unroll
            for (int j = 0; j < 8; ++j) { f[j] = lo[j]; f[j + 8] = hi[j]; }
            bf[i] = f;
        }

        v8f acc = {};
#pragma unroll
        for (int i = 0; i < 8; ++i) {
            const unsigned short* ap = a_lds + m * A_STRIDE + akoff + i * 32;
            v8bf alo = *(const v8bf*)(ap);                 // K = i*32+akoff+0..7
            v8bf ahi = *(const v8bf*)(ap + 8);             // K = i*32+akoff+8..15
            v16bf af;
#pragma unroll
            for (int j = 0; j < 8; ++j) { af[j] = alo[j]; af[j + 8] = ahi[j]; }
            acc = __builtin_amdgcn_wmma_f32_16x16x32_bf16(
                      false, af, false, bf[i], (short)0, acc, false, false);
        }

        // dist(m, code) = ||e||^2 - 2*dot  (||x||^2 is row-constant -> irrelevant for argmin)
        // codes scanned in ascending order => strict '<' keeps the first (lowest) index.
        const int code = n0 + l16;
#pragma unroll
        for (int r = 0; r < 8; ++r) {
            float d = __builtin_fmaf(-2.0f, acc[r], en[tile]);
            if (d < best[r]) { best[r] = d; bidx[r] = code; }
        }
    }

    // ---- butterfly argmin inside each 16-lane half (rows r / r+8) ----
#pragma unroll
    for (int r = 0; r < 8; ++r) {
        float v  = best[r];
        int   ix = bidx[r];
#pragma unroll
        for (int s = 1; s < 16; s <<= 1) {
            float ov = __shfl_xor(v, s, 32);
            int   oi = __shfl_xor(ix, s, 32);
            if (ov < v || (ov == v && oi < ix)) { v = ov; ix = oi; }
        }
        best[r] = v; bidx[r] = ix;
    }
    if (l16 == 0) {   // lane 0 -> rows 0..7, lane 16 -> rows 8..15
#pragma unroll
        for (int r = 0; r < 8; ++r) {
            red_val[r + half * 8][wave] = best[r];
            red_idx[r + half * 8][wave] = bidx[r];
        }
    }
    __syncthreads();

    // ---- cross-wave argmin, emit indices ----
    if (t < 16) {
        float v  = red_val[t][0];
        int   ix = red_idx[t][0];
#pragma unroll
        for (int w = 1; w < NWAVE; ++w) {
            float ov = red_val[t][w];
            int   oi = red_idx[t][w];
            if (ov < v || (ov == v && oi < ix)) { v = ov; ix = oi; }
        }
        final_idx[t]       = ix;
        out_idx[nbase + t] = ix;
    }
    __syncthreads();

    // ---- gather codebook rows (fp32), write NCHW output, accumulate loss ----
    float lsum = 0.0f;
    {
        const int row = t & 15;
        const int c0  = t >> 4;
        const int idx = final_idx[row];
#pragma unroll
        for (int i = 0; i < 16; ++i) {
            int   c  = c0 + 16 * i;
            float q  = embed[c * K_CODES + idx];       // L2-resident gather
            float xv = x_lds[row * X_STRIDE + c];
            float d  = q - xv;
            lsum     = __builtin_fmaf(d, d, lsum);
            out_q[xbase + c * HW + row] = q;           // straight-through fwd value = q
        }
    }
#pragma unroll
    for (int s = 16; s >= 1; s >>= 1) lsum += __shfl_xor(lsum, s, 32);
    if (lane == 0) lred[wave] = lsum;
    __syncthreads();
    if (t == 0) {
        float tot = 0.0f;
#pragma unroll
        for (int w = 0; w < NWAVE; ++w) tot += lred[w];
        loss_partial[blockIdx.x] = tot;
    }
}

// ---------------- Stage 2: deterministic loss reduction ----------------
__global__ __launch_bounds__(256) void vq_loss_final(const float* __restrict__ partial,
                                                     float* __restrict__ loss_out) {
    __shared__ float lred[NWAVE];
    const int t = threadIdx.x;
    float s = 0.0f;
    for (int i = t; i < 4096; i += 256) s += partial[i];
#pragma unroll
    for (int sh = 16; sh >= 1; sh >>= 1) s += __shfl_xor(s, sh, 32);
    if ((t & 31) == 0) lred[t >> 5] = s;
    __syncthreads();
    if (t == 0) {
        float tot = 0.0f;
#pragma unroll
        for (int w = 0; w < NWAVE; ++w) tot += lred[w];
        // q_latent_loss + COMMITMENT * e_latent_loss = (1 + 0.25) * mse
        loss_out[0] = 1.25f * tot / 16777216.0f;      // mean over B*H*W*C
    }
}

extern "C" void kernel_launch(void* const* d_in, const int* in_sizes, int n_in,
                              void* d_out, int out_size, void* d_ws, size_t ws_size,
                              hipStream_t stream) {
    const float* x     = (const float*)d_in[0];   // [64,256,32,32] f32
    const float* embed = (const float*)d_in[1];   // [256,1024] f32

    // workspace layout: bf16 embT (512 KB) | enorm (4 KB) | loss partials (16 KB)
    unsigned short* embT   = (unsigned short*)d_ws;
    float*          enorm  = (float*)((char*)d_ws + (size_t)K_CODES * C_DIM * sizeof(unsigned short));
    float*          part   = enorm + K_CODES;

    // output layout: quantize [16777216] f32 | loss [1] f32 | embed_idx [65536] i32
    float* out_q    = (float*)d_out;
    float* loss_out = out_q + (size_t)N_ROWS * C_DIM;
    int*   out_idx  = (int*)(loss_out + 1);

    vq_prep_embT <<<128, 256, 0, stream>>>(embed, embT);
    vq_prep_enorm<<<K_CODES / 256, 256, 0, stream>>>(embed, enorm);
    vq_stage1    <<<N_ROWS / 16, 256, 0, stream>>>(x, embed, embT, enorm, out_q, out_idx, part);
    vq_loss_final<<<1, 256, 0, stream>>>(part, loss_out);
}
